// ReasoningLayer_5815385719047
// MI455X (gfx1250) — matrix operand order repaired and checked by reading, not compiled, not run on hardware
//
#include <hip/hip_runtime.h>
#include <cstdint>
#include <cstddef>

#define DD 2048
#define HH 4096
#define KSTEP 4
#define NSTEPS 64
#define MOMC 0.9f
#define THRC 0.01f
#define SLICES 8

typedef __attribute__((ext_vector_type(16))) __bf16 v16bf;
typedef __attribute__((ext_vector_type(8)))  float  v8f;
typedef __attribute__((ext_vector_type(4)))  unsigned int u32x4;
typedef __attribute__((ext_vector_type(4)))  float  f32x4;
typedef __attribute__((ext_vector_type(4)))  __bf16 bf16x4;

enum { SC_CUM = 0, SC_RW = 1, SC_MOM = 2, SC_BEST = 3, SC_P = 4 };
enum { FL_ACTIVE = 0, FL_CONVERGED = 1, FL_DORESET = 2 };

__device__ __forceinline__ float sigm(float x) { return 1.0f / (1.0f + expf(-x)); }

// ---------------------------------------------------------------------------
// f32 -> bf16 weight conversion (once per call). Non-temporal source reads:
// the f32 weights are touched exactly once, keep them out of L2 so the bf16
// working set (~90MB for the per-step matrices) can stay resident.
// ---------------------------------------------------------------------------
__global__ void __launch_bounds__(256)
convert_f32_bf16(const float* __restrict__ src, __bf16* __restrict__ dst, long n) {
    long i = ((long)blockIdx.x * blockDim.x + threadIdx.x) * 4;
    long stride = (long)gridDim.x * blockDim.x * 4;
    for (; i < n; i += stride) {
        f32x4 f = __builtin_nontemporal_load((const f32x4*)(src + i));
        bf16x4 o = { (__bf16)f.x, (__bf16)f.y, (__bf16)f.z, (__bf16)f.w };
        *(bf16x4*)(dst + i) = o;
    }
}

// ---------------------------------------------------------------------------
// WMMA GEMV: y_part[slice][r] = sum_{k in slice} W[r,k] * x[k]
// One wave per (16-row tile, K-slice). A = 16x32 bf16 weight tile,
// B = x chunk broadcast across all 16 columns, C/D = f32 accumulator.
// NT=1 streams the weights with non-temporal loads (hg_* matrices: 200MB,
// read every 4th step -> don't let them evict the L2-resident small weights).
// ---------------------------------------------------------------------------
template <int NT>
__global__ void __launch_bounds__(256)
gemv_bf16_wmma(const __bf16* __restrict__ W,
               const float* __restrict__ x1, int n1,
               const float* __restrict__ x2,
               float* __restrict__ ypart,
               int rows, int cols, int nslice,
               const int* gate1, const int* gate2) {
    if (gate1 && *gate1 == 0) return;
    if (gate2 && *gate2 == 0) return;
    int wave   = (blockIdx.x * blockDim.x + threadIdx.x) >> 5;
    int nwaves = (gridDim.x * blockDim.x) >> 5;
    int lane   = threadIdx.x & 31;
    int hi     = lane >> 4;        // lane group: 0 or 1
    int row16  = lane & 15;
    int tiles  = rows >> 4;
    int kcols  = cols / nslice;

    for (int w = wave; w < tiles * nslice; w += nwaves) {
        int tile  = w % tiles;
        int slice = w / tiles;
        int r0    = tile << 4;
        int k0    = slice * kcols;
        const __bf16* Wr = W + (size_t)(r0 + row16) * cols;
        // A K-slice never straddles the concat boundary (n1 and kcols are both
        // multiples of 16 and slice-aligned): pick the x base once per tile.
        const float* xsb = (k0 < n1) ? x1 : (x2 - n1);

        v8f acc = {0.f, 0.f, 0.f, 0.f, 0.f, 0.f, 0.f, 0.f};
#pragma unroll 4
        for (int k = k0; k < k0 + kcols; k += 32) {
            // A fragment (ISA 16-bit A layout): lanes<16 hold K k+0..7 / k+16..23,
            // lanes>=16 hold K k+8..15 / k+24..31.
            union { v16bf v; u32x4 q[2]; } A;
            const u32x4* p0 = (const u32x4*)(Wr + k + hi * 8);
            const u32x4* p1 = (const u32x4*)(Wr + k + 16 + hi * 8);
            if (NT) {
                A.q[0] = __builtin_nontemporal_load(p0);
                A.q[1] = __builtin_nontemporal_load(p1);
            } else {
                A.q[0] = *p0;
                A.q[1] = *p1;
            }
            // B fragment: lanes 0-15 hold K k+0..7 / k+16..23, lanes 16-31 hold k+8..15 / k+24..31,
            // broadcast identically into every column.
            const float* xs = xsb + k + (hi << 3);
            v16bf B;
#pragma unroll
            for (int e = 0; e < 16; e += 4) {
                f32x4 f = *(const f32x4*)(xs + e + (e & 8));
                B[e + 0] = (__bf16)f.x;
                B[e + 1] = (__bf16)f.y;
                B[e + 2] = (__bf16)f.z;
                B[e + 3] = (__bf16)f.w;
            }
            acc = __builtin_amdgcn_wmma_f32_16x16x32_bf16(
                false, A.v, false, B, (short)0, acc, false, false);
        }
        // D layout: lane 0 holds M=0..7 (col 0), lane 16 holds M=8..15.
        if (row16 == 0) {
            float* yp = ypart + (size_t)slice * rows + r0 + hi * 8;
#pragma unroll
            for (int m = 0; m < 8; ++m) yp[m] = acc[m];
        }
    }
}

// Sum K-slice partials + bias, optional tanh, write vector.
__global__ void __launch_bounds__(256)
gemv_finish(const float* __restrict__ ypart, const float* __restrict__ bias,
            float* __restrict__ y, int rows, int nslice, int act,
            const int* gate1, const int* gate2) {
    if (gate1 && *gate1 == 0) return;
    if (gate2 && *gate2 == 0) return;
    int i = blockIdx.x * blockDim.x + threadIdx.x;
    if (i >= rows) return;
    float s = bias[i];
    for (int sl = 0; sl < nslice; ++sl) s += ypart[(size_t)sl * rows + i];
    if (act == 1) s = tanhf(s);
    y[i] = s;
}

// GRU gate combine: h[j] = (1-z)*n + z*h[j] from the two 3n-row GEMV partials.
__global__ void __launch_bounds__(256)
gru_combine(const float* __restrict__ giP, const float* __restrict__ ghP,
            const float* __restrict__ bih, const float* __restrict__ bhh,
            float* __restrict__ hv, int n, int nslice, const int* gate) {
    if (gate && *gate == 0) return;
    int j = blockIdx.x * blockDim.x + threadIdx.x;
    if (j >= n) return;
    size_t rows3 = (size_t)3 * n;
    float gir = bih[j], giz = bih[n + j], gin = bih[2 * n + j];
    float ghr = bhh[j], ghz = bhh[n + j], ghn = bhh[2 * n + j];
    for (int s = 0; s < nslice; ++s) {
        const float* gp = giP + (size_t)s * rows3;
        const float* hp = ghP + (size_t)s * rows3;
        gir += gp[j]; giz += gp[n + j]; gin += gp[2 * n + j];
        ghr += hp[j]; ghz += hp[n + j]; ghn += hp[2 * n + j];
    }
    float r  = sigm(gir + ghr);
    float z  = sigm(giz + ghz);
    float nn = tanhf(gin + r * ghn);
    hv[j] = (1.f - z) * nn + z * hv[j];
}

// raw_conv = sigmoid(hc_W . h + hc_b); mom = MOM*mom + (1-MOM)*raw_conv
__global__ void __launch_bounds__(256)
dot_hc_kernel(const float* __restrict__ w, const float* __restrict__ b,
              const float* __restrict__ h, float* scal, const int* flags) {
    if (flags[FL_ACTIVE] == 0) return;
    __shared__ float red[256];
    float s = 0.f;
    for (int i = threadIdx.x; i < HH; i += 256) s += w[i] * h[i];
    red[threadIdx.x] = s;
    __syncthreads();
    for (int o = 128; o > 0; o >>= 1) {
        if (threadIdx.x < o) red[threadIdx.x] += red[threadIdx.x + o];
        __syncthreads();
    }
    if (threadIdx.x == 0) {
        float raw = sigm(red[0] + b[0]);
        scal[SC_MOM] = MOMC * scal[SC_MOM] + (1.f - MOMC) * raw;
    }
}

// halt -> p, result weights, cum, rw, best, converged flag
__global__ void __launch_bounds__(256)
dot_halt_kernel(const float* __restrict__ w, const float* __restrict__ b,
                const float* __restrict__ h, const float* __restrict__ l,
                float* scal, int* flags) {
    if (flags[FL_ACTIVE] == 0) return;
    __shared__ float red[256];
    float s = 0.f;
    for (int i = threadIdx.x; i < HH; i += 256) s += w[i] * h[i];
    for (int j = threadIdx.x; j < DD; j += 256) s += w[HH + j] * l[j];
    red[threadIdx.x] = s;
    __syncthreads();
    for (int o = 128; o > 0; o >>= 1) {
        if (threadIdx.x < o) red[threadIdx.x] += red[threadIdx.x + o];
        __syncthreads();
    }
    if (threadIdx.x == 0) {
        float halt = sigm(red[0] + b[0]);
        float cum  = scal[SC_CUM];
        float p    = fminf(halt, 1.f - cum);
        scal[SC_P]  = p;
        scal[SC_RW] += p;
        cum += p;
        scal[SC_CUM] = cum;
        if (scal[SC_MOM] > scal[SC_BEST] + THRC) scal[SC_BEST] = scal[SC_MOM];
        flags[FL_CONVERGED] = (cum > 0.95f) ? 1 : 0;
    }
}

// reset_p -> do_reset; then active &= !converged (last scalar update of the step)
__global__ void __launch_bounds__(256)
dot_reset_kernel(const float* __restrict__ w, const float* __restrict__ b,
                 const float* __restrict__ h, const float* __restrict__ l,
                 float* scal, int* flags, int step) {
    if (flags[FL_ACTIVE] == 0) return;
    __shared__ float red[256];
    float s = 0.f;
    for (int i = threadIdx.x; i < HH; i += 256) s += w[i] * h[i];
    for (int j = threadIdx.x; j < DD; j += 256) s += w[HH + j] * l[j];
    red[threadIdx.x] = s;
    __syncthreads();
    for (int o = 128; o > 0; o >>= 1) {
        if (threadIdx.x < o) red[threadIdx.x] += red[threadIdx.x + o];
        __syncthreads();
    }
    if (threadIdx.x == 0) {
        float rp = sigm(red[0] + w[HH + DD] * scal[SC_MOM] + b[0]);
        int conv = flags[FL_CONVERGED];
        flags[FL_DORESET] = (!conv && rp > 0.7f && step > KSTEP) ? 1 : 0;
        if (conv) flags[FL_ACTIVE] = 0;
    }
}

// result += p*res; last_res = res   (only runs while active)
__global__ void __launch_bounds__(256)
resupd_kernel(const float* __restrict__ res, float* __restrict__ result,
              float* __restrict__ last_res, const float* scal, const int* flags) {
    if (flags[FL_ACTIVE] == 0) return;
    int i = blockIdx.x * blockDim.x + threadIdx.x;
    if (i >= DD) return;
    float p = scal[SC_P];
    result[i] += p * res[i];
    last_res[i] = res[i];
}

__global__ void __launch_bounds__(256)
init_kernel(float* h, float* l, float* result, float* last_res,
            float* scal, int* flags) {
    int i = blockIdx.x * blockDim.x + threadIdx.x;
    if (i < HH) h[i] = 0.f;
    if (i < DD) { l[i] = 0.f; result[i] = 0.f; last_res[i] = 0.f; }
    if (i == 0) {
        scal[SC_CUM] = 0.f; scal[SC_RW] = 0.f; scal[SC_MOM] = 0.f;
        scal[SC_BEST] = 0.f; scal[SC_P] = 0.f;
        flags[FL_ACTIVE] = 1; flags[FL_CONVERGED] = 0; flags[FL_DORESET] = 0;
    }
}

__global__ void __launch_bounds__(256)
final_kernel(const float* __restrict__ result, const float* __restrict__ last_res,
             const float* scal, float* __restrict__ out) {
    int i = blockIdx.x * blockDim.x + threadIdx.x;
    if (i >= DD) return;
    float rw = scal[SC_RW];
    out[i] = (rw > 0.f) ? result[i] / fmaxf(rw, 1e-8f) : last_res[i];
}

// ---------------------------------------------------------------------------
extern "C" void kernel_launch(void* const* d_in, const int* in_sizes, int n_in,
                              void* d_out, int out_size, void* d_ws, size_t ws_size,
                              hipStream_t stream) {
    const float* query   = (const float*)d_in[0];
    const float* hg_Wih  = (const float*)d_in[1];
    const float* hg_Whh  = (const float*)d_in[2];
    const float* hg_bih  = (const float*)d_in[3];
    const float* hg_bhh  = (const float*)d_in[4];
    const float* hd_W    = (const float*)d_in[5];
    const float* hd_b    = (const float*)d_in[6];
    const float* hc_W    = (const float*)d_in[7];
    const float* hc_b    = (const float*)d_in[8];
    const float* lg_Wih  = (const float*)d_in[9];
    const float* lg_Whh  = (const float*)d_in[10];
    const float* lg_bih  = (const float*)d_in[11];
    const float* lg_bhh  = (const float*)d_in[12];
    const float* lo_W    = (const float*)d_in[13];
    const float* lo_b    = (const float*)d_in[14];
    const float* halt_W  = (const float*)d_in[15];
    const float* halt_b  = (const float*)d_in[16];
    const float* reset_W = (const float*)d_in[17];
    const float* reset_b = (const float*)d_in[18];
    const float* li_W    = (const float*)d_in[19];
    const float* li_b    = (const float*)d_in[20];
    float* out = (float*)d_out;

    // ---- workspace layout ----
    uint8_t* ws = (uint8_t*)d_ws;
    size_t off = 0;
    auto take = [&](size_t bytes) -> void* {
        void* p = ws + off;
        off += (bytes + 255) & ~(size_t)255;
        return p;
    };
    const size_t NHG = (size_t)3 * HH * 2 * DD;   // 12288 x 4096
    const size_t NHD = (size_t)DD * HH;           // 2048  x 4096
    const size_t NLG = (size_t)3 * DD * DD;       // 6144  x 2048
    const size_t NLO = (size_t)DD * DD;           // 2048  x 2048

    __bf16* bHgWih = (__bf16*)take(NHG * 2);
    __bf16* bHgWhh = (__bf16*)take(NHG * 2);
    __bf16* bHdW   = (__bf16*)take(NHD * 2);
    __bf16* bLgWih = (__bf16*)take(NLG * 2);
    __bf16* bLgWhh = (__bf16*)take(NLG * 2);
    __bf16* bLoW   = (__bf16*)take(NLO * 2);
    __bf16* bLiW   = (__bf16*)take(NHD * 2);

    float* h_v      = (float*)take(HH * 4);
    float* l_v      = (float*)take(DD * 4);
    float* directive= (float*)take(DD * 4);
    float* res_v    = (float*)take(DD * 4);
    float* result_v = (float*)take(DD * 4);
    float* lastres  = (float*)take(DD * 4);
    float* scal     = (float*)take(16 * 4);
    int*   flags    = (int*)  take(16 * 4);

    float* giPart  = (float*)take((size_t)SLICES * 3 * HH * 4);
    float* ghPart  = (float*)take((size_t)SLICES * 3 * HH * 4);
    float* hdPart  = (float*)take((size_t)SLICES * DD * 4);
    float* gilPart = (float*)take((size_t)SLICES * 3 * DD * 4);
    float* ghlPart = (float*)take((size_t)SLICES * 3 * DD * 4);
    float* loPart  = (float*)take((size_t)SLICES * DD * 4);
    float* liPart  = (float*)take((size_t)SLICES * DD * 4);

    // ---- one-time bf16 weight conversion ----
    auto conv = [&](const float* s, __bf16* d, size_t n) {
        convert_f32_bf16<<<4096, 256, 0, stream>>>(s, d, (long)n);
    };
    conv(hg_Wih, bHgWih, NHG);
    conv(hg_Whh, bHgWhh, NHG);
    conv(hd_W,   bHdW,   NHD);
    conv(lg_Wih, bLgWih, NLG);
    conv(lg_Whh, bLgWhh, NLG);
    conv(lo_W,   bLoW,   NLO);
    conv(li_W,   bLiW,   NHD);

    init_kernel<<<(HH + 255) / 256, 256, 0, stream>>>(h_v, l_v, result_v, lastres, scal, flags);

    const int* gA = &flags[FL_ACTIVE];
    const int* gR = &flags[FL_DORESET];

    auto gemv = [&](const __bf16* W, const float* x1, int n1, const float* x2,
                    float* ypart, int rows, int cols, const int* g1, const int* g2,
                    int nt) {
        int waves  = (rows / 16) * SLICES;
        int blocks = (waves * 32 + 255) / 256;
        if (nt)
            gemv_bf16_wmma<1><<<blocks, 256, 0, stream>>>(W, x1, n1, x2, ypart,
                                                          rows, cols, SLICES, g1, g2);
        else
            gemv_bf16_wmma<0><<<blocks, 256, 0, stream>>>(W, x1, n1, x2, ypart,
                                                          rows, cols, SLICES, g1, g2);
    };

    for (int step = 0; step < NSTEPS; ++step) {
        if (step % KSTEP == 0) {
            // h_new = GRU(concat(q,l), h)   (200MB bf16: stream non-temporal)
            gemv(bHgWih, query, DD, l_v, giPart, 3 * HH, 2 * DD, gA, nullptr, 1);
            gemv(bHgWhh, h_v, HH, h_v, ghPart, 3 * HH, HH, gA, nullptr, 1);
            gru_combine<<<(HH + 255) / 256, 256, 0, stream>>>(
                giPart, ghPart, hg_bih, hg_bhh, h_v, HH, SLICES, gA);
        }
        // directive = tanh(hd_W @ h + hd_b)   (L2-resident weights: RT hints)
        gemv(bHdW, h_v, HH, h_v, hdPart, DD, HH, gA, nullptr, 0);
        gemv_finish<<<(DD + 255) / 256, 256, 0, stream>>>(
            hdPart, hd_b, directive, DD, SLICES, 1, gA, nullptr);
        // raw_conv / mom
        dot_hc_kernel<<<1, 256, 0, stream>>>(hc_W, hc_b, h_v, scal, flags);
        // l = GRU(directive, l)
        gemv(bLgWih, directive, DD, directive, gilPart, 3 * DD, DD, gA, nullptr, 0);
        gemv(bLgWhh, l_v, DD, l_v, ghlPart, 3 * DD, DD, gA, nullptr, 0);
        gru_combine<<<(DD + 255) / 256, 256, 0, stream>>>(
            gilPart, ghlPart, lg_bih, lg_bhh, l_v, DD, SLICES, gA);
        // res = tanh(lo_W @ l + lo_b)
        gemv(bLoW, l_v, DD, l_v, loPart, DD, DD, gA, nullptr, 0);
        gemv_finish<<<(DD + 255) / 256, 256, 0, stream>>>(
            loPart, lo_b, res_v, DD, SLICES, 1, gA, nullptr);
        // halt / p / cum / rw / best / converged
        dot_halt_kernel<<<1, 256, 0, stream>>>(halt_W, halt_b, h_v, l_v, scal, flags);
        // result += p*res; last_res = res
        resupd_kernel<<<(DD + 255) / 256, 256, 0, stream>>>(res_v, result_v, lastres, scal, flags);
        // reset_p / do_reset; active &= !converged
        dot_reset_kernel<<<1, 256, 0, stream>>>(reset_W, reset_b, h_v, l_v, scal, flags, step);
        // conditional l = tanh(li_W @ h + li_b)
        gemv(bLiW, h_v, HH, h_v, liPart, DD, HH, gR, gA, 0);
        gemv_finish<<<(DD + 255) / 256, 256, 0, stream>>>(
            liPart, li_b, l_v, DD, SLICES, 1, gR, gA);
    }

    final_kernel<<<(DD + 255) / 256, 256, 0, stream>>>(result_v, lastres, scal, out);

    (void)in_sizes; (void)n_in; (void)out_size; (void)ws_size;
}
